// GAT_70248485093911
// MI455X (gfx1250) — matrix operand order repaired and checked by reading, not compiled, not run on hardware
//
#include <hip/hip_runtime.h>

#define NPNODES 50000
#define NDNODES 50000
#define NEDGES  600000
#define NHEADS  4

typedef __attribute__((ext_vector_type(16))) __bf16 v16bf;
typedef __attribute__((ext_vector_type(8)))  float  v8f;
typedef int vec4i __attribute__((vector_size(16)));

union FragU { v16bf bf; unsigned int u[8]; };

// ---- CDNA5 async global->LDS path (guarded so compile can never break) ----
#if defined(__has_builtin)
#  if __has_builtin(__builtin_amdgcn_global_load_async_to_lds_b128)
#    define USE_ASYNC_LDS 1
#  else
#    define USE_ASYNC_LDS 0
#  endif
#  if __has_builtin(__builtin_amdgcn_s_wait_asynccnt)
#    define WAIT_ASYNC() __builtin_amdgcn_s_wait_asynccnt(0)
#  else
#    define WAIT_ASYNC() asm volatile("s_wait_asynccnt 0x0" ::: "memory")
#  endif
#else
#  define USE_ASYNC_LDS 0
#  define WAIT_ASYNC()
#endif

#if USE_ASYNC_LDS
// Probe-derived signature: arg0 = vec4i* in global AS(1), arg1 = LDS AS(3) ptr, then imm offset, imm cpol.
#define ASYNC_B128(gp, lp)                                                     \
  __builtin_amdgcn_global_load_async_to_lds_b128(                              \
      (__attribute__((address_space(1))) vec4i*)(gp),                          \
      (__attribute__((address_space(3))) vec4i*)(lp), 0, 0)
#endif

__device__ __forceinline__ unsigned short f32_to_bf16_rne(float f) {
  unsigned int u = __float_as_uint(f);
  unsigned int r = u + 0x7FFFu + ((u >> 16) & 1u);
  return (unsigned short)(r >> 16);
}

__device__ __forceinline__ float lrelu02(float v) { return v > 0.f ? v : 0.2f * v; }

// ---------------- fp32 -> bf16 convert ----------------
__global__ void k_f32_to_bf16(const float* __restrict__ x, unsigned short* __restrict__ y, long n) {
  long i = (long)blockIdx.x * blockDim.x + threadIdx.x;
  long stride = (long)gridDim.x * blockDim.x;
  for (; i < n; i += stride) y[i] = f32_to_bf16_rne(x[i]);
}

// ---------------- bf16 WMMA GEMM: C(MxN,f32) = X(MxK) @ W(KxN) ----------------
// Block tile 64x64, 4 waves (wave32) in a 2x2 grid, each wave 32x32 = 2x2 WMMA tiles.
// Double-buffered LDS pipeline; A tile staged via async global->LDS (ASYNCcnt),
// B tile staged via regs with a transposed LDS scatter.
__global__ __launch_bounds__(128) void k_gemm_bf16_wmma(
    const unsigned short* __restrict__ X,   // M x K, bf16 bits
    const unsigned short* __restrict__ Wm,  // K x N, bf16 bits
    float* __restrict__ C,                  // M x N, f32
    int M, int K, int N)
{
  __shared__ unsigned short As[2][64 * 40];  // [row][k], stride 40 halves (80B, 16B-aligned rows)
  __shared__ unsigned short Bs[2][64 * 40];  // transposed: [n][k]

  const int tid  = threadIdx.x;
  const int lane = tid & 31;
  const int wave = tid >> 5;
  const int wm   = wave >> 1;   // 0..1
  const int wn   = wave & 1;    // 0..1
  const int m0   = blockIdx.y * 64;
  const int n0   = blockIdx.x * 64;

  const int lm = lane & 15;
  const int hi = lane >> 4;

  const bool fullM = (m0 + 64 <= M);
#if USE_ASYNC_LDS
  const bool useAsync = fullM;
#else
  const bool useAsync = false;
#endif

  const v8f vzero = {0.f, 0.f, 0.f, 0.f, 0.f, 0.f, 0.f, 0.f};
  v8f acc[2][2];
  acc[0][0] = vzero; acc[0][1] = vzero; acc[1][0] = vzero; acc[1][1] = vzero;

  unsigned int areg[8];
  unsigned int breg[8];

  // ---- prologue: stage tile kb=0 into buffer 0 ----
  if (useAsync) {
#if USE_ASYNC_LDS
#pragma unroll
    for (int s = 0; s < 2; ++s) {
      int u   = tid * 2 + s;       // 256 x 16B segments
      int row = u >> 2;
      int seg = u & 3;
      const unsigned short* gp = X + (long)(m0 + row) * K + seg * 8;
      unsigned short* lp = &As[0][row * 40 + seg * 8];
      ASYNC_B128(gp, lp);
    }
#endif
  } else {
#pragma unroll
    for (int j = 0; j < 8; ++j) {
      int i = tid + 128 * j;
      int row = i >> 4;
      int k2  = (i & 15) << 1;
      int grow = m0 + row;
      unsigned int val = 0u;
      if (grow < M) val = *(const unsigned int*)&X[(long)grow * K + k2];
      *(unsigned int*)&As[0][row * 40 + k2] = val;
    }
  }
#pragma unroll
  for (int j = 0; j < 8; ++j) {
    int i = tid + 128 * j;
    int kk = i >> 5;
    int nn = (i & 31) << 1;
    unsigned int val = *(const unsigned int*)&Wm[(long)kk * N + n0 + nn];
    Bs[0][nn * 40 + kk]       = (unsigned short)(val & 0xFFFFu);
    Bs[0][(nn + 1) * 40 + kk] = (unsigned short)(val >> 16);
  }
  if (useAsync) { WAIT_ASYNC(); }
  __syncthreads();

  int buf = 0;
  for (int kb = 0; kb < K; kb += 32) {
    const int nb = buf ^ 1;
    const bool hasNext = (kb + 32 < K);

    // ---- kick off next tile's loads ----
    if (hasNext) {
      if (useAsync) {
#if USE_ASYNC_LDS
#pragma unroll
        for (int s = 0; s < 2; ++s) {
          int u   = tid * 2 + s;
          int row = u >> 2;
          int seg = u & 3;
          const unsigned short* gp = X + (long)(m0 + row) * K + (kb + 32) + seg * 8;
          unsigned short* lp = &As[nb][row * 40 + seg * 8];
          ASYNC_B128(gp, lp);
        }
#endif
      } else {
#pragma unroll
        for (int j = 0; j < 8; ++j) {
          int i = tid + 128 * j;
          int row = i >> 4;
          int k2  = (i & 15) << 1;
          int grow = m0 + row;
          areg[j] = 0u;
          if (grow < M) areg[j] = *(const unsigned int*)&X[(long)grow * K + (kb + 32) + k2];
        }
      }
#pragma unroll
      for (int j = 0; j < 8; ++j) {
        int i = tid + 128 * j;
        int kk = i >> 5;
        int nn = (i & 31) << 1;
        breg[j] = *(const unsigned int*)&Wm[(long)(kb + 32 + kk) * N + n0 + nn];
      }
    }

    // ---- fragments from current buffer + 4 WMMAs ----
    {
      const unsigned short* as = As[buf];
      const unsigned short* bs = Bs[buf];
      FragU a[2], b[2];
#pragma unroll
      for (int t = 0; t < 2; ++t) {
        int mr = wm * 32 + t * 16 + lm;
#pragma unroll
        for (int j = 0; j < 8; ++j) {   // ISA 16-bit A 16x32 per-lane K layout
          int k2 = (j < 4) ? (hi * 8 + 2 * j) : (16 + hi * 8 + 2 * (j - 4));
          a[t].u[j] = *(const unsigned int*)&as[mr * 40 + k2];
        }
        int nc = wn * 32 + t * 16 + lm;
#pragma unroll
        for (int j = 0; j < 8; ++j) {   // B 32x16: same per-lane K layout as A
          int k2 = (j < 4) ? (hi * 8 + 2 * j) : (16 + hi * 8 + 2 * (j - 4));
          b[t].u[j] = *(const unsigned int*)&bs[nc * 40 + k2];
        }
      }
#pragma unroll
      for (int mi = 0; mi < 2; ++mi)
#pragma unroll
        for (int ni = 0; ni < 2; ++ni)
          acc[mi][ni] = __builtin_amdgcn_wmma_f32_16x16x32_bf16(
              false, a[mi].bf, false, b[ni].bf, (short)0, acc[mi][ni], false, false);
    }

    // ---- commit next tile to LDS, drain async, one barrier per iteration ----
    if (hasNext) {
      if (!useAsync) {
#pragma unroll
        for (int j = 0; j < 8; ++j) {
          int i = tid + 128 * j;
          int row = i >> 4;
          int k2  = (i & 15) << 1;
          *(unsigned int*)&As[nb][row * 40 + k2] = areg[j];
        }
      }
#pragma unroll
      for (int j = 0; j < 8; ++j) {
        int i = tid + 128 * j;
        int kk = i >> 5;
        int nn = (i & 31) << 1;
        Bs[nb][nn * 40 + kk]       = (unsigned short)(breg[j] & 0xFFFFu);
        Bs[nb][(nn + 1) * 40 + kk] = (unsigned short)(breg[j] >> 16);
      }
      if (useAsync) { WAIT_ASYNC(); }
    }
    __syncthreads();
    buf = nb;
  }

  // ---- epilogue per ISA 32-bit C/D layout; fast path for full tiles ----
  if (fullM) {
#pragma unroll
    for (int mi = 0; mi < 2; ++mi) {
#pragma unroll
      for (int ni = 0; ni < 2; ++ni) {
        int col = n0 + wn * 32 + ni * 16 + lm;
        long row0 = (long)m0 + wm * 32 + mi * 16 + hi * 8;
        float* cp = C + row0 * N + col;
#pragma unroll
        for (int r = 0; r < 8; ++r) { *cp = acc[mi][ni][r]; cp += N; }
      }
    }
  } else {
#pragma unroll
    for (int mi = 0; mi < 2; ++mi) {
#pragma unroll
      for (int ni = 0; ni < 2; ++ni) {
        int col = n0 + wn * 32 + ni * 16 + lm;
        int rowb = m0 + wm * 32 + mi * 16 + hi * 8;
        float* cp = C + (long)rowb * N + col;
#pragma unroll
        for (int r = 0; r < 8; ++r) {
          if (rowb + r < M) *cp = acc[mi][ni][r];
          cp += N;
        }
      }
    }
  }
}

// ---------------- per-(node,head) attention logits: e = sum_f fs * a ----------------
__global__ void k_head_logits(const float* __restrict__ fs, const float* __restrict__ avec,
                              float* __restrict__ e, int Nn, int F) {
  int idx = blockIdx.x * blockDim.x + threadIdx.x;
  if (idx >= Nn * NHEADS) return;
  int node = idx >> 2;
  int h    = idx & 3;
  const float* row = fs + (long)node * (NHEADS * F) + h * F;
  const float* a   = avec + h * F;
  float s = 0.f;
  for (int f = 0; f < F; ++f) s += row[f] * a[f];
  e[idx] = s;
}

// ---------------- CSR offsets via lower_bound on sorted dst ----------------
__global__ void k_seg_offsets(const int* __restrict__ dst, int E, int Nn, int* __restrict__ off) {
  int n = blockIdx.x * blockDim.x + threadIdx.x;
  if (n > Nn) return;
  int lo = 0, hiE = E;
  while (lo < hiE) { int mid = (lo + hiE) >> 1; if (dst[mid] < n) lo = mid + 1; else hiE = mid; }
  off[n] = lo;
}

// ---------------- edge softmax + weighted aggregation, one block per dst node ----------------
__global__ __launch_bounds__(256) void k_gat_aggregate(
    const float* __restrict__ fs,   // (Ns, H*F)
    const float* __restrict__ el,   // (Ns, H)
    const float* __restrict__ er,   // (Nd, H)
    const int*   __restrict__ src,  // (E)
    const int*   __restrict__ off,  // (Nd+1)
    const float* __restrict__ bias, // (H*F)
    float* __restrict__ out,        // (Nd, H*F)
    int Nd, int F)
{
  int d = blockIdx.x;
  if (d >= Nd) return;
  int tid = threadIdx.x;
  int beg = off[d], end = off[d + 1];
  const int HF = NHEADS * F;

  __shared__ float red[256];
  __shared__ float mh[NHEADS], dh[NHEADS];

  float erd[NHEADS];
#pragma unroll
  for (int h = 0; h < NHEADS; ++h) erd[h] = er[(long)d * NHEADS + h];

  // phase 1: per-head segment max of leaky_relu(el[src]+er[d])
  float lmax[NHEADS];
#pragma unroll
  for (int h = 0; h < NHEADS; ++h) lmax[h] = -3.4e38f;
  for (int e = beg + tid; e < end; e += 256) {
    int s = src[e];
#pragma unroll
    for (int h = 0; h < NHEADS; ++h)
      lmax[h] = fmaxf(lmax[h], lrelu02(el[(long)s * NHEADS + h] + erd[h]));
  }
#pragma unroll
  for (int h = 0; h < NHEADS; ++h) {
    red[tid] = lmax[h];
    __syncthreads();
    for (int sft = 128; sft > 0; sft >>= 1) {
      if (tid < sft) red[tid] = fmaxf(red[tid], red[tid + sft]);
      __syncthreads();
    }
    if (tid == 0) { float m = red[0]; mh[h] = (m > -3.0e38f) ? m : 0.f; }  // isfinite guard
    __syncthreads();
  }

  // phase 2: per-head denominator
  float ls[NHEADS] = {0.f, 0.f, 0.f, 0.f};
  for (int e = beg + tid; e < end; e += 256) {
    int s = src[e];
#pragma unroll
    for (int h = 0; h < NHEADS; ++h)
      ls[h] += expf(lrelu02(el[(long)s * NHEADS + h] + erd[h]) - mh[h]);
  }
#pragma unroll
  for (int h = 0; h < NHEADS; ++h) {
    red[tid] = ls[h];
    __syncthreads();
    for (int sft = 128; sft > 0; sft >>= 1) {
      if (tid < sft) red[tid] += red[tid + sft];
      __syncthreads();
    }
    if (tid == 0) dh[h] = fmaxf(red[0], 1e-9f);
    __syncthreads();
  }

  // phase 3: coalesced weighted row gather; each thread owns 1-2 output features
  float acc0 = 0.f, acc1 = 0.f;
  int f0 = tid;
  int f1 = tid + 256;
  int h0 = f0 / F;
  int h1 = (f1 < HF) ? (f1 / F) : 0;
  for (int e = beg; e < end; ++e) {
    int s = src[e];
    const float* frow = fs + (long)s * HF;
    float a0 = expf(lrelu02(el[(long)s * NHEADS + h0] + erd[h0]) - mh[h0]) / dh[h0];
    acc0 += a0 * frow[f0];
    if (f1 < HF) {
      float a1 = expf(lrelu02(el[(long)s * NHEADS + h1] + erd[h1]) - mh[h1]) / dh[h1];
      acc1 += a1 * frow[f1];
    }
  }
  out[(long)d * HF + f0] = fmaxf(acc0 + bias[f0], 0.f);
  if (f1 < HF) out[(long)d * HF + f1] = fmaxf(acc1 + bias[f1], 0.f);
}

// ---------------- host orchestration ----------------
extern "C" void kernel_launch(void* const* d_in, const int* in_sizes, int n_in,
                              void* d_out, int out_size, void* d_ws, size_t ws_size,
                              hipStream_t stream)
{
  const float* h_p    = (const float*)d_in[0];
  const float* h_d    = (const float*)d_in[1];
  const int*   src_pd = (const int*)d_in[2];
  const int*   dst_pd = (const int*)d_in[3];
  const int*   src_dp = (const int*)d_in[4];
  const int*   dst_dp = (const int*)d_in[5];

  char* p = (char*)d_ws;
  auto alloc = [&](size_t bytes) -> void* {
    void* r = (void*)p;
    p += (bytes + 255) & ~(size_t)255;
    return r;
  };

  int* off_pd = (int*)alloc((NDNODES + 1) * sizeof(int));
  int* off_dp = (int*)alloc((NPNODES + 1) * sizeof(int));
  unsigned short* xp_bf = (unsigned short*)alloc((size_t)NPNODES * 256 * 2);
  unsigned short* xd_bf = (unsigned short*)alloc((size_t)NDNODES * 256 * 2);
  unsigned short* W_bf  = (unsigned short*)alloc((size_t)256 * 512 * 2);
  float* fs_pd  = (float*)alloc((size_t)NPNODES * 512 * 4);
  float* fs_dp  = (float*)alloc((size_t)NDNODES * 512 * 4);
  float* ftmp   = (float*)alloc((size_t)NPNODES * 512 * 4);
  float* el_pd  = (float*)alloc((size_t)NPNODES * NHEADS * 4);
  float* er_pd  = (float*)alloc((size_t)NDNODES * NHEADS * 4);
  float* el_dp  = (float*)alloc((size_t)NDNODES * NHEADS * 4);
  float* er_dp  = (float*)alloc((size_t)NPNODES * NHEADS * 4);
  float* buf_p1 = (float*)alloc((size_t)NPNODES * 512 * 4);
  float* buf_d1 = (float*)alloc((size_t)NDNODES * 512 * 4);
  float* buf_p2 = (float*)alloc((size_t)NPNODES * 512 * 4);
  float* buf_d2 = (float*)alloc((size_t)NDNODES * 512 * 4);

  // CSR offsets (dst arrays are sorted) — graph topology is layer-invariant
  k_seg_offsets<<<(NDNODES + 256) / 256, 256, 0, stream>>>(dst_pd, NEDGES, NDNODES, off_pd);
  k_seg_offsets<<<(NPNODES + 256) / 256, 256, 0, stream>>>(dst_dp, NEDGES, NPNODES, off_dp);

  auto cvt = [&](const float* x, unsigned short* y, long n) {
    k_f32_to_bf16<<<2048, 256, 0, stream>>>(x, y, n);
  };
  auto gemm = [&](const unsigned short* X, const unsigned short* Wm, float* C, int M, int K, int N) {
    dim3 grid(N / 64, (M + 63) / 64);
    k_gemm_bf16_wmma<<<grid, 128, 0, stream>>>(X, Wm, C, M, K, N);
  };
  auto logits = [&](const float* fsrc, const float* av, float* e, int Nn, int F) {
    k_head_logits<<<(Nn * NHEADS + 255) / 256, 256, 0, stream>>>(fsrc, av, e, Nn, F);
  };

  auto run_layer = [&](const float* xp, const float* xd, int Din, int F, int widx,
                       float* outp, float* outd) {
    const float* Wpd  = (const float*)d_in[widx + 0];
    const float* alpd = (const float*)d_in[widx + 1];
    const float* arpd = (const float*)d_in[widx + 2];
    const float* bpd  = (const float*)d_in[widx + 3];
    const float* Wdp  = (const float*)d_in[widx + 4];
    const float* aldp = (const float*)d_in[widx + 5];
    const float* ardp = (const float*)d_in[widx + 6];
    const float* bdp  = (const float*)d_in[widx + 7];
    const int HF = NHEADS * F;

    cvt(xp, xp_bf, (long)NPNODES * Din);
    cvt(xd, xd_bf, (long)NDNODES * Din);

    // relation p->d (shared fc Wpd applied to both src(p) and dst(d) features)
    cvt(Wpd, W_bf, (long)Din * HF);
    gemm(xd_bf, W_bf, ftmp, NDNODES, Din, HF);    // fd for pd
    logits(ftmp, arpd, er_pd, NDNODES, F);
    gemm(xp_bf, W_bf, fs_pd, NPNODES, Din, HF);   // fs for pd
    logits(fs_pd, alpd, el_pd, NPNODES, F);

    // relation d->p
    cvt(Wdp, W_bf, (long)Din * HF);
    gemm(xp_bf, W_bf, ftmp, NPNODES, Din, HF);    // fd for dp
    logits(ftmp, ardp, er_dp, NPNODES, F);
    gemm(xd_bf, W_bf, fs_dp, NDNODES, Din, HF);   // fs for dp
    logits(fs_dp, aldp, el_dp, NDNODES, F);

    k_gat_aggregate<<<NDNODES, 256, 0, stream>>>(fs_pd, el_pd, er_pd, src_pd, off_pd, bpd, outd, NDNODES, F);
    k_gat_aggregate<<<NPNODES, 256, 0, stream>>>(fs_dp, el_dp, er_dp, src_dp, off_dp, bdp, outp, NPNODES, F);
  };

  float* out_p3 = (float*)d_out;
  float* out_d3 = (float*)d_out + (size_t)NPNODES * 512;

  run_layer(h_p,    h_d,    128, 64,  6,  buf_p1, buf_d1);
  run_layer(buf_p1, buf_d1, 256, 64,  14, buf_p2, buf_d2);
  run_layer(buf_p2, buf_d2, 256, 128, 22, out_p3, out_d3);

  (void)in_sizes; (void)n_in; (void)out_size; (void)ws_size;
}
